// ConceptFlow_43782896616134
// MI455X (gfx1250) — matrix operand order repaired and checked by reading, not compiled, not run on hardware
//
#include <hip/hip_runtime.h>

// ---------------- CDNA5 WMMA types / helpers ----------------
typedef __attribute__((ext_vector_type(16))) __bf16 v16bf;
typedef __attribute__((ext_vector_type(8)))  float  v8f;

__device__ inline __bf16 f2bf(float x) {
  unsigned u = __builtin_bit_cast(unsigned, x);
  unsigned r = (u + 0x7FFFu + ((u >> 16) & 1u)) >> 16;   // RNE truncate
  return __builtin_bit_cast(__bf16, (unsigned short)r);
}

__device__ inline v8f wmma_bf16(v16bf a, v16bf b, v8f c) {
  return __builtin_amdgcn_wmma_f32_16x16x32_bf16(false, a, false, b, (short)0, c,
                                                 false, false);
}

// Packed fragment layout: tile (T, kT) of a matrix, 32 lanes x 16 bf16 contiguous
// per lane (32B). One v16bf load per fragment -> 2x global_load_b128.
__device__ inline v16bf ld_frag(const __bf16* __restrict__ P, int nKT,
                                int T, int kT, int lane) {
  return *(const v16bf*)(P + ((long)(T * nKT + kT) * 32 + lane) * 16);
}

// Scatter one (row,col) f32 value into A-fragment packed layout.
// A layout (ISA 7.12.2): lane&15=M; K = (i&7) + ((i>>3)<<4) + ((lane>>4)<<3).
__device__ inline void st_packed(__bf16* __restrict__ P, int nKT,
                                 int row, int col, float v) {
  int mT = row >> 4, kT = col >> 5, kl = col & 31;
  int lane = (row & 15) | (((kl >> 3) & 1) << 4);
  int idx  = ((kl >> 4) << 3) | (kl & 7);
  P[((long)(mT * nKT + kT) * 32 + lane) * 16 + idx] = f2bf(v);
}

__device__ inline float sigf(float x) { return 1.0f / (1.0f + __expf(-x)); }

// ---------------- one-time packers ---------------------------------------------------
// Weights W[N,K] row-major f32 -> B-fragment packed bf16 (lane&15=N; K=(i&7)+((i>>3)<<4)+((lane>>4)<<3)),
// zero-padded to 16x32 tile multiples. Clamp+select (no predicated loads).
__global__ void pack_w(const float* __restrict__ W, __bf16* __restrict__ P,
                       int N, int K, int ldw, int nKT) {
  int lane = threadIdx.x & 31, wave = threadIdx.x >> 5;
  int nT = blockIdx.x;
  int kT = blockIdx.y * 4 + wave;
  if (kT >= nKT) return;
  int n  = nT * 16 + (lane & 15);
  int kb = kT * 32 + ((lane >> 4) << 3);
  const float* row = W + (long)(n < N ? n : N - 1) * ldw;
  bool nok = n < N;
  v16bf f;
#pragma unroll
  for (int i = 0; i < 16; ++i) {
    int k = kb + (i & 7) + ((i >> 3) << 4);
    float x = row[k < K ? k : K - 1];
    f[i] = f2bf((nok && k < K) ? x : 0.0f);
  }
  *(v16bf*)(P + ((long)(nT * nKT + kT) * 32 + lane) * 16) = f;
}

// Activations A[M,K] f32 (exact tile multiples) -> A-fragment packed bf16.
__global__ void pack_a(const float* __restrict__ A, __bf16* __restrict__ P,
                       int lda, int nKT) {
  int lane = threadIdx.x & 31, wave = threadIdx.x >> 5;
  int mT = blockIdx.x;
  int kT = blockIdx.y * 4 + wave;
  if (kT >= nKT) return;
  int m  = mT * 16 + (lane & 15);
  int kb = kT * 32 + ((lane >> 4) << 3);
  const float* row = A + (long)m * lda + kb;
  v16bf f;
#pragma unroll
  for (int i = 0; i < 16; ++i) f[i] = f2bf(row[(i & 7) + ((i >> 3) << 4)]);
  *(v16bf*)(P + ((long)(mT * nKT + kT) * 32 + lane) * 16) = f;
}

// ---------------- GEMM on packed operands: out = A*W^T (+bias) ----------------------
// grid (ceil(nNT/4), M/16), block 128 (4 waves, one 16x16 tile each).
// Optional outputs: f32 row-major, and up to two packed-A-layout copies.
__global__ void gemm_pp(const __bf16* __restrict__ Ap, const __bf16* __restrict__ Bp,
                        const float* __restrict__ bias, int N, int nNT, int KT,
                        float* __restrict__ outF, int ldo,
                        __bf16* __restrict__ outP1, int p1KT,
                        __bf16* __restrict__ outP2, int p2KT) {
  int lane = threadIdx.x & 31, wave = threadIdx.x >> 5;
  int nT = blockIdx.x * 4 + wave;
  int mT = blockIdx.y;
  if (nT >= nNT) return;                  // wave-uniform
  v8f acc = {};
  for (int kT = 0; kT < KT; ++kT) {
    v16bf fa = ld_frag(Ap, KT, mT, kT, lane);
    v16bf fb = ld_frag(Bp, KT, nT, kT, lane);
    acc = wmma_bf16(fa, fb, acc);
  }
  int n  = nT * 16 + (lane & 15);
  int mb = mT * 16 + ((lane >> 4) << 3);  // C layout: VGPR v -> M = mb+v, N = lane&15
  if (n < N) {
    float bv = bias ? bias[n] : 0.0f;
#pragma unroll
    for (int v = 0; v < 8; ++v) {
      float val = acc[v] + bv;
      int m = mb + v;
      if (outF)  outF[(long)m * ldo + n] = val;
      if (outP1) st_packed(outP1, p1KT, m, n, val);
      if (outP2) st_packed(outP2, p2KT, m, n, val);
    }
  }
}

// ---------------- GRU recurrent step: fused 3-gate WMMA + nonlinear update ----------
// hP packed [32,512]; WhP packed [1536,512]; GI = x@Wi^T+bi f32 [B,S,1536].
// grid 16 x 128; wave w -> (mT=w>>1, u-subtile w&1).
__global__ void gru_step(const __bf16* __restrict__ hP, const float* __restrict__ hF,
                         const __bf16* __restrict__ WhP, const float* __restrict__ bh,
                         const float* __restrict__ GI, int t,
                         __bf16* __restrict__ hPout, float* __restrict__ hFout,
                         __bf16* __restrict__ yPout) {
  int lane = threadIdx.x & 31, wave = threadIdx.x >> 5;
  int mT = wave >> 1;
  int u0 = blockIdx.x * 32 + (wave & 1) * 16;
  int nTr = u0 >> 4, nTz = (512 + u0) >> 4, nTn = (1024 + u0) >> 4;
  v8f ar = {}, az = {}, an = {};
  for (int kT = 0; kT < 16; ++kT) {
    v16bf fa = ld_frag(hP, 16, mT, kT, lane);
    ar = wmma_bf16(fa, ld_frag(WhP, 16, nTr, kT, lane), ar);
    az = wmma_bf16(fa, ld_frag(WhP, 16, nTz, kT, lane), az);
    an = wmma_bf16(fa, ld_frag(WhP, 16, nTn, kT, lane), an);
  }
  int u  = u0 + (lane & 15);
  int mb = mT * 16 + ((lane >> 4) << 3);
  float bhr = bh[u], bhz = bh[512 + u], bhn = bh[1024 + u];
#pragma unroll
  for (int v = 0; v < 8; ++v) {
    int b = mb + v;
    long g = ((long)b * 128 + t) * 1536;
    float r  = sigf(GI[g + u] + ar[v] + bhr);
    float z  = sigf(GI[g + 512 + u] + az[v] + bhz);
    float nn = tanhf(GI[g + 1024 + u] + r * (an[v] + bhn));   // n = tanh(i_n + r*(h Wh_n + bh_n))
    float hold = hF[(long)b * 512 + u];
    float hnew = (1.0f - z) * nn + z * hold;
    hFout[(long)b * 512 + u] = hnew;
    st_packed(hPout, 16, b, u, hnew);
    if (yPout) st_packed(yPout, 16, b * 128 + t, u, hnew);
  }
}

// ---------------- LSTM gates: g = x@Wi^T + h@Wh^T + bi + bh (all packed) ------------
// grid (32, 32), block 128
__global__ void lstm_gates(const __bf16* __restrict__ xP, const __bf16* __restrict__ hP,
                           const __bf16* __restrict__ WiP, const __bf16* __restrict__ WhP,
                           const float* __restrict__ bi, const float* __restrict__ bh,
                           float* __restrict__ g) {
  int lane = threadIdx.x & 31, wave = threadIdx.x >> 5;
  int nT = blockIdx.x * 4 + wave;   // 0..127
  int mT = blockIdx.y;              // 0..31
  v8f acc = {};
  for (int kT = 0; kT < 16; ++kT) {
    acc = wmma_bf16(ld_frag(xP, 16, mT, kT, lane), ld_frag(WiP, 16, nT, kT, lane), acc);
    acc = wmma_bf16(ld_frag(hP, 16, mT, kT, lane), ld_frag(WhP, 16, nT, kT, lane), acc);
  }
  int n  = nT * 16 + (lane & 15);
  int mb = mT * 16 + ((lane >> 4) << 3);
  float bv = bi[n] + bh[n];
#pragma unroll
  for (int v = 0; v < 8; ++v) g[(long)(mb + v) * 2048 + n] = acc[v] + bv;
}

// ---------------- LSTM elementwise update (writes packed h for next WMMA) -----------
__global__ void lstm_update(const float* __restrict__ g, __bf16* __restrict__ hP,
                            float* __restrict__ c, float* __restrict__ Hdst, int hstride) {
  int m = blockIdx.x, u = threadIdx.x;
  const float* gr = g + (long)m * 2048;
  float i  = sigf(gr[u]);
  float f  = sigf(gr[512 + u]);
  float gg = tanhf(gr[1024 + u]);
  float o  = sigf(gr[1536 + u]);
  float cc = f * c[(long)m * 512 + u] + i * gg;
  c[(long)m * 512 + u] = cc;
  float hh = o * tanhf(cc);
  st_packed(hP, 16, m, u, hh);
  if (Hdst) Hdst[(long)m * hstride + u] = hh;
}

// ---------------- gathers / init -----------------------------------------------------
__global__ void zero_k(float* p, int n) {
  int i = blockIdx.x * 256 + threadIdx.x;
  if (i < n) p[i] = 0.0f;
}

__global__ void gather_text(const int* __restrict__ q, const float* __restrict__ wemb,
                            __bf16* __restrict__ XeP) {        // packed [4096, 320]
  int r = blockIdx.x;
  long w = q[r];
  for (int j = threadIdx.x; j < 320; j += blockDim.x) {
    float x = (j < 300) ? wemb[w * 300 + j] : 0.0f;
    st_packed(XeP, 10, r, j, x);
  }
}

__global__ void gather_e(const int* __restrict__ gi, const float* __restrict__ ent,
                         int t, __bf16* __restrict__ catP) {   // cols 512..639 of cat
  int bp = blockIdx.x;
  long cand = gi[bp * 384 + t * 64];           // [BP, L=6, C=64], candidate 0
  for (int j = threadIdx.x; j < 128; j += blockDim.x) {
    float x = (j < 100) ? ent[cand * 100 + j] : 0.0f;
    st_packed(catP, 20, bp, 512 + j, x);
  }
}

__global__ void gc0_init(const float* __restrict__ tf, __bf16* __restrict__ gcP,
                         __bf16* __restrict__ catP, float* __restrict__ H) {
  int bp = blockIdx.x;
  for (int u = threadIdx.x; u < 512; u += blockDim.x) {
    float v = tf[(long)(bp >> 4) * 512 + u];
    st_packed(gcP, 16, bp, u, v);
    st_packed(catP, 20, bp, u, v);
    H[(long)bp * 3072 + u] = v;
  }
}

// ---------------- candidate scoring + masked BCE (deterministic) --------------------
__global__ void score_bce(const float* __restrict__ Wbuf, const float* __restrict__ ent,
                          const int* __restrict__ gi, const float* __restrict__ bias,
                          const int* __restrict__ pnum, const int* __restrict__ plen,
                          float* __restrict__ ploss, float* __restrict__ pcnt) {
  int blk = blockIdx.x;             // bp*6 + l
  int bp = blk / 6, l = blk % 6;
  int b = bp >> 4, p = bp & 15;
  __shared__ float wrow[100];
  __shared__ float red[64];
  for (int j = threadIdx.x; j < 100; j += 64) wrow[j] = Wbuf[(long)blk * 100 + j];
  __syncthreads();
  int c = threadIdx.x;
  long cand = gi[bp * 384 + l * 64 + c];
  const float* er = ent + cand * 100;
  float dot = 0.0f;
  for (int j = 0; j < 100; ++j) dot += er[j] * wrow[j];
  float logit = dot + bias[0];
  bool valid = (p < pnum[b]) && (l < plen[b * 16 + p]);
  float target = (valid && c == 0) ? 1.0f : 0.0f;
  float bce = fmaxf(logit, 0.0f) - logit * target + log1pf(__expf(-fabsf(logit)));
  red[c] = valid ? bce : 0.0f;
  __syncthreads();
  for (int s = 32; s > 0; s >>= 1) {
    if (c < s) red[c] += red[c + s];
    __syncthreads();
  }
  if (c == 0) { ploss[blk] = red[0]; pcnt[blk] = valid ? 64.0f : 0.0f; }
}

__global__ void final_reduce(const float* __restrict__ ploss,
                             const float* __restrict__ pcnt, float* __restrict__ out) {
  __shared__ float sl[256], sc[256];
  float a = 0.0f, n = 0.0f;
  for (int i = threadIdx.x; i < 3072; i += 256) { a += ploss[i]; n += pcnt[i]; }
  sl[threadIdx.x] = a; sc[threadIdx.x] = n;
  __syncthreads();
  for (int s = 128; s > 0; s >>= 1) {
    if (threadIdx.x < s) { sl[threadIdx.x] += sl[threadIdx.x + s];
                           sc[threadIdx.x] += sc[threadIdx.x + s]; }
    __syncthreads();
  }
  if (threadIdx.x == 0) out[0] = sl[0] / sc[0];
}

// ---------------- host orchestration -------------------------------------------------
extern "C" void kernel_launch(void* const* d_in, const int* in_sizes, int n_in,
                              void* d_out, int out_size, void* d_ws, size_t ws_size,
                              hipStream_t stream) {
  if (n_in < 27) return;
  const float *Wc, *Wp, *bc, *bias, *ent, *wemb;
  const float *gWh[2], *gWi[2], *gbh[2], *gbi[2];
  const float *lWh[2], *lWi[2], *lbh[2], *lbi[2];
  auto F = [&](int i) { return (const float*)d_in[i]; };
  auto I = [&](int i) { return (const int*)d_in[i]; };
  if (in_sizes[0] == 512 * 612) {   // JAX sorted-key pytree flatten
    Wc = F(0); Wp = F(1); bc = F(2); bias = F(3); ent = F(4);
    gWh[0] = F(5);  gWi[0] = F(6);  gbh[0] = F(7);  gbi[0] = F(8);
    gWh[1] = F(9);  gWi[1] = F(10); gbh[1] = F(11); gbi[1] = F(12);
    lWh[0] = F(13); lWi[0] = F(14); lbh[0] = F(15); lbi[0] = F(16);
    lWh[1] = F(17); lWi[1] = F(18); lbh[1] = F(19); lbi[1] = F(20);
    wemb = F(21);
  } else {                          // insertion-order flatten
    wemb = F(0); ent = F(1);
    gWi[0] = F(2);  gWh[0] = F(3);  gbi[0] = F(4);  gbh[0] = F(5);
    gWi[1] = F(6);  gWh[1] = F(7);  gbi[1] = F(8);  gbh[1] = F(9);
    lWi[0] = F(10); lWh[0] = F(11); lbi[0] = F(12); lbh[0] = F(13);
    lWi[1] = F(14); lWh[1] = F(15); lbi[1] = F(16); lbh[1] = F(17);
    Wc = F(18); bc = F(19); Wp = F(20); bias = F(21);
  }
  const int* qtext  = I(22);
  const int* ginput = I(23);
  const int* pnum   = I(25);
  const int* plen   = I(26);

  // ---- scratch layout (float units; all offsets 32B-aligned) ----
  float* ws = (float*)d_ws;
  float*  GI    = ws;                         // 4096*1536 f32 (both GRU layers)
  __bf16* XeP   = (__bf16*)(GI + 6291456);    // 4096*320 bf16   (655360 f)
  __bf16* Y1P   = (__bf16*)(GI + 6291456 + 655360);             // 4096*512 bf16 (1048576 f)
  float*  hFA   = GI + 6291456 + 655360 + 1048576;              // 32*512
  float*  hFB   = hFA + 16384;
  __bf16* hPA   = (__bf16*)(hFB + 16384);     // 32*512 bf16 (8192 f)
  __bf16* hPB   = (__bf16*)(hFB + 16384 + 8192);
  __bf16* gcP   = (__bf16*)(hFB + 16384 + 16384);               // 512*512 bf16 (131072 f)
  __bf16* catP  = (__bf16*)(hFB + 16384 + 16384 + 131072);      // 512*640 bf16 (163840 f)
  float*  gates = hFB + 16384 + 16384 + 131072 + 163840;        // 512*2048
  __bf16* h1P   = (__bf16*)(gates + 1048576);                   // 512*512 bf16 (131072 f)
  __bf16* h2P   = (__bf16*)(gates + 1048576 + 131072);
  float*  c1    = gates + 1048576 + 262144;                     // 512*512
  float*  c2    = c1 + 262144;
  float*  H     = c2 + 262144;                                  // 512*6*512
  __bf16* HP    = (__bf16*)(H + 1572864);                       // 3072*512 bf16 (786432 f)
  float*  Wbuf  = H + 1572864 + 786432;                         // 3072*100
  float*  ploss = Wbuf + 307200;                                // 3072
  float*  pcnt  = ploss + 3072;                                 // 3072
  float*  wp0   = pcnt + 3072;                                  // packed weights below
  __bf16* PgWi0 = (__bf16*)wp0;               wp0 += 245760;    // 1536 x 320
  __bf16* PgWh0 = (__bf16*)wp0;               wp0 += 393216;    // 1536 x 512
  __bf16* PgWi1 = (__bf16*)wp0;               wp0 += 393216;    // 1536 x 512
  __bf16* PgWh1 = (__bf16*)wp0;               wp0 += 393216;    // 1536 x 512
  __bf16* PlWi0 = (__bf16*)wp0;               wp0 += 524288;    // 2048 x 512
  __bf16* PlWh0 = (__bf16*)wp0;               wp0 += 524288;
  __bf16* PlWi1 = (__bf16*)wp0;               wp0 += 524288;
  __bf16* PlWh1 = (__bf16*)wp0;               wp0 += 524288;
  __bf16* PWc   = (__bf16*)wp0;               wp0 += 163840;    // 512 x 640
  __bf16* PWp   = (__bf16*)wp0;               wp0 += 28672;     // 112 x 512
  (void)ws_size; (void)out_size;

  // ---- pack weights (once per launch) ----
  pack_w<<<dim3(96, 3),  128, 0, stream>>>(gWi[0], PgWi0, 1536, 300, 300, 10);
  pack_w<<<dim3(96, 4),  128, 0, stream>>>(gWh[0], PgWh0, 1536, 512, 512, 16);
  pack_w<<<dim3(96, 4),  128, 0, stream>>>(gWi[1], PgWi1, 1536, 512, 512, 16);
  pack_w<<<dim3(96, 4),  128, 0, stream>>>(gWh[1], PgWh1, 1536, 512, 512, 16);
  pack_w<<<dim3(128, 4), 128, 0, stream>>>(lWi[0], PlWi0, 2048, 512, 512, 16);
  pack_w<<<dim3(128, 4), 128, 0, stream>>>(lWh[0], PlWh0, 2048, 512, 512, 16);
  pack_w<<<dim3(128, 4), 128, 0, stream>>>(lWi[1], PlWi1, 2048, 512, 512, 16);
  pack_w<<<dim3(128, 4), 128, 0, stream>>>(lWh[1], PlWh1, 2048, 512, 512, 16);
  pack_w<<<dim3(32, 5),  128, 0, stream>>>(Wc, PWc, 512, 612, 612, 20);
  pack_w<<<dim3(7, 4),   128, 0, stream>>>(Wp, PWp, 100, 512, 512, 16);

  // ---- text encoder ----
  gather_text<<<4096, 128, 0, stream>>>(qtext, wemb, XeP);
  // GI1 = Xe @ Wi1^T + bi1 (batched over all timesteps): [4096x320] x [320x1536]
  gemm_pp<<<dim3(24, 256), 128, 0, stream>>>(XeP, PgWi0, gbi[0], 1536, 96, 10,
                                             GI, 1536, nullptr, 0, nullptr, 0);
  zero_k<<<96, 256, 0, stream>>>(hFA, 16384 + 16384 + 8192);     // hFA + first half of hFB only (96*256 = 24576 f)
  {
    float* hf = hFA; float* hf2 = hFB; __bf16* hp = hPA; __bf16* hp2 = hPB;
    for (int t = 0; t < 128; ++t) {
      gru_step<<<16, 128, 0, stream>>>(hp, hf, PgWh0, gbh[0], GI, t, hp2, hf2, Y1P);
      float* tf_ = hf; hf = hf2; hf2 = tf_;
      __bf16* tp = hp; hp = hp2; hp2 = tp;
    }
    // GI2 = Y1 @ Wi2^T + bi2 (reuse GI)
    gemm_pp<<<dim3(24, 256), 128, 0, stream>>>(Y1P, PgWi1, gbi[1], 1536, 96, 16,
                                               GI, 1536, nullptr, 0, nullptr, 0);
    // zero hf2 and first half (4096 f) of hp2 as layer-2 initial state
    zero_k<<<64, 256, 0, stream>>>(hf2, 16384);
    zero_k<<<16, 256, 0, stream>>>((float*)hp2, 4096);
    __bf16* cp = hp2; float* cf = hf2; __bf16* np = hp; float* nf = hf;
    for (int t = 0; t < 128; ++t) {
      gru_step<<<16, 128, 0, stream>>>(cp, cf, PgWh1, gbh[1], GI, t, np, nf, nullptr);
      float* tf_ = cf; cf = nf; nf = tf_;
      __bf16* tp = cp; cp = np; np = tp;
    }
    // text_final = cf ; seed decoder context + H row 0
    gc0_init<<<512, 128, 0, stream>>>(cf, gcP, catP, H);
  }

  // ---- graph decoder ----
  zero_k<<<3072, 256, 0, stream>>>(gates + 1048576, 786432);     // h1P,h2P,c1,c2
  for (int t = 0; t < 5; ++t) {
    gather_e<<<512, 128, 0, stream>>>(ginput, ent, t, catP);
    // gc = cat @ Wc^T + bc  -> written into gates-A packed and cat packed cols 0..511
    gemm_pp<<<dim3(8, 32), 128, 0, stream>>>(catP, PWc, bc, 512, 32, 20,
                                             nullptr, 0, gcP, 16, catP, 20);
    lstm_gates<<<dim3(32, 32), 128, 0, stream>>>(gcP, h1P, PlWi0, PlWh0,
                                                 lbi[0], lbh[0], gates);
    lstm_update<<<512, 512, 0, stream>>>(gates, h1P, c1, nullptr, 0);
    lstm_gates<<<dim3(32, 32), 128, 0, stream>>>(h1P, h2P, PlWi1, PlWh1,
                                                 lbi[1], lbh[1], gates);
    lstm_update<<<512, 512, 0, stream>>>(gates, h2P, c2, H + (t + 1) * 512, 3072);
  }

  // ---- scoring + loss ----
  pack_a<<<dim3(192, 4), 128, 0, stream>>>(H, HP, 512, 16);
  gemm_pp<<<dim3(2, 192), 128, 0, stream>>>(HP, PWp, nullptr, 100, 7, 16,
                                            Wbuf, 100, nullptr, 0, nullptr, 0);
  score_bce<<<3072, 64, 0, stream>>>(Wbuf, ent, ginput, bias, pnum, plen, ploss, pcnt);
  final_reduce<<<1, 256, 0, stream>>>(ploss, pcnt, (float*)d_out);
}
